// GAT_68882685493578
// MI455X (gfx1250) — matrix operand order repaired and checked by reading, not compiled, not run on hardware
//
#include <hip/hip_runtime.h>
#include <hip/hip_bf16.h>

typedef __bf16 bf16;
typedef __attribute__((ext_vector_type(16))) __bf16 v16bf;
typedef __attribute__((ext_vector_type(8)))  __bf16 v8bf;
typedef __attribute__((ext_vector_type(8)))  float  v8f;

#define LEAKY 0.2f

// ---------------------------------------------------------------- utilities

__global__ void k_fill_f32(float* __restrict__ p, float v, long long n) {
    long long i = (long long)blockIdx.x * blockDim.x + threadIdx.x;
    if (i < n) p[i] = v;
}

__global__ void k_f32_to_bf16(const float* __restrict__ in, bf16* __restrict__ out, long long n) {
    long long i = (long long)blockIdx.x * blockDim.x + threadIdx.x;
    if (i < n) out[i] = (bf16)in[i];
}

// Pack W (K x N, row-major f32) into B-fragment order for v_wmma_f32_16x16x32_bf16.
// Fragment block (nblk, kblk) covers N[nblk*16, +16), K[kblk*32, +32).
// Per lane l: n = nblk*16 + (l&15), K = kblk*32 + (l<16 ? 0 : 8) + (j<8 ? j : j+8), j=0..15,
// stored contiguously: Wp[((nblk*KB + kblk)*32 + l)*16 + j].
__global__ void k_pack_w(const float* __restrict__ W, bf16* __restrict__ Wp, int K, int N) {
    long long idx = (long long)blockIdx.x * blockDim.x + threadIdx.x;
    long long total = (long long)K * N;
    if (idx >= total) return;
    int j    = (int)(idx & 15);
    int lane = (int)((idx >> 4) & 31);
    long long blk = idx >> 9;            // nblk*KB + kblk
    int KB = K >> 5;
    int nblk = (int)(blk / KB);
    int kblk = (int)(blk % KB);
    int n = nblk * 16 + (lane & 15);
    int k = kblk * 32 + ((lane < 16) ? 0 : 8) + ((j < 8) ? j : j + 8);
    Wp[idx] = (bf16)W[(long long)k * N + n];
}

// --------------------------------------------------------------- WMMA GEMM
// C[M,N] = A[M,K] (bf16 row-major) * Wp (pre-packed bf16 B fragments), f32 out.
// One wave computes a 16x64 tile: 1 A fragment reused across 4 B fragments per k-step.
__global__ void k_gemm_wmma_bf16(const bf16* __restrict__ A, const bf16* __restrict__ Wp,
                                 float* __restrict__ C, int M, int N, int K) {
    const int lane  = threadIdx.x & 31;
    const int wave  = threadIdx.x >> 5;
    const int wpb   = blockDim.x >> 5;
    const int tilesN = N >> 6;                 // 64-wide wave tiles
    const int tilesM = M >> 4;
    const int tile  = blockIdx.x * wpb + wave;
    if (tile >= tilesM * tilesN) return;
    const int tm = tile / tilesN;
    const int tn = tile % tilesN;
    const int m0 = tm << 4;
    const int n0 = tn << 6;
    const int KB = K >> 5;

    const int mrow   = m0 + (lane & 15);
    const int kshift = (lane < 16) ? 0 : 8;    // ISA 16-bit A-matrix 16x32 layout
    const bf16* arow = A + (long long)mrow * K;

    v8f acc[4] = {v8f{}, v8f{}, v8f{}, v8f{}};

    for (int kb = 0; kb < KB; ++kb) {
        const int kbase = (kb << 5) + kshift;
        v8bf alo = *reinterpret_cast<const v8bf*>(arow + kbase);
        v8bf ahi = *reinterpret_cast<const v8bf*>(arow + kbase + 16);
        v16bf afrag = __builtin_shufflevector(alo, ahi,
            0,1,2,3,4,5,6,7,8,9,10,11,12,13,14,15);
#pragma unroll
        for (int j = 0; j < 4; ++j) {
            const long long nblk = (n0 >> 4) + j;
            const v16bf bfrag = *reinterpret_cast<const v16bf*>(
                Wp + ((nblk * KB + kb) * 32 + lane) * 16);
            acc[j] = __builtin_amdgcn_wmma_f32_16x16x32_bf16(
                false, afrag, false, bfrag, (short)0, acc[j], false, false);
        }
    }

    // C/D layout: VGPR r -> row (lane<16 ? r : r+8), col = lane&15
    const int crow = m0 + ((lane < 16) ? 0 : 8);
    const int ccol = lane & 15;
#pragma unroll
    for (int j = 0; j < 4; ++j) {
#pragma unroll
        for (int r = 0; r < 8; ++r) {
            C[(long long)(crow + r) * N + n0 + (j << 4) + ccol] = acc[j][r];
        }
    }
}

// ------------------------------------------------------------ GAT edge phase

// el[n,h] = feat[n,h,:].al[h,:],  er likewise
__global__ void k_attn_dots(const float* __restrict__ feat,
                            const float* __restrict__ al, const float* __restrict__ ar,
                            float* __restrict__ el, float* __restrict__ er,
                            int Nn, int H, int D) {
    int idx = blockIdx.x * blockDim.x + threadIdx.x;   // n*H + h
    if (idx >= Nn * H) return;
    int h = idx % H, n = idx / H;
    const float* f = feat + (long long)n * H * D + (long long)h * D;
    const float* a = al + h * D;
    const float* b = ar + h * D;
    float sl = 0.f, sr = 0.f;
    for (int d = 0; d < D; ++d) { float v = f[d]; sl += v * a[d]; sr += v * b[d]; }
    el[idx] = sl; er[idx] = sr;
}

__device__ __forceinline__ void atomicMaxF(float* addr, float val) {
    int* ai = (int*)addr;
    int old = __float_as_int(*addr);
    while (__int_as_float(old) < val) {
        int assumed = old;
        old = atomicCAS(ai, assumed, __float_as_int(val));
        if (old == assumed) break;
    }
}

__global__ void k_edge_logits(const int* __restrict__ src, const int* __restrict__ dst,
                              const float* __restrict__ el, const float* __restrict__ er,
                              float* __restrict__ e, float* __restrict__ emax, int E, int H) {
    int idx = blockIdx.x * blockDim.x + threadIdx.x;   // edge*H + h
    if (idx >= E * H) return;
    int h = idx % H, ed = idx / H;
    int s = src[ed], d0 = dst[ed];
    float v = el[s * H + h] + er[d0 * H + h];
    v = (v > 0.f) ? v : LEAKY * v;
    e[idx] = v;
    atomicMaxF(&emax[d0 * H + h], v);
}

__global__ void k_edge_exp(const int* __restrict__ dst, const float* __restrict__ emax,
                           float* __restrict__ e, float* __restrict__ denom, int E, int H) {
    int idx = blockIdx.x * blockDim.x + threadIdx.x;
    if (idx >= E * H) return;
    int h = idx % H, ed = idx / H;
    int d0 = dst[ed];
    float ex = __expf(e[idx] - emax[d0 * H + h]);
    e[idx] = ex;
    atomicAdd(&denom[d0 * H + h], ex);
}

__global__ void k_edge_scatter(const int* __restrict__ src, const int* __restrict__ dst,
                               const float* __restrict__ feat, const float* __restrict__ ex,
                               const float* __restrict__ denom, float* __restrict__ rst,
                               int E, int H, int D) {
    long long idx = (long long)blockIdx.x * blockDim.x + threadIdx.x; // edge*(H*D) + f
    const int HD = H * D;
    if (idx >= (long long)E * HD) return;
    int f  = (int)(idx % HD);
    int ed = (int)(idx / HD);
    int h  = f / D;
    int s = src[ed], d0 = dst[ed];
    float alpha = ex[(long long)ed * H + h] / denom[d0 * H + h];
    atomicAdd(&rst[(long long)d0 * HD + f], feat[(long long)s * HD + f] * alpha);
}

__global__ void k_bias_relu_bf16(const float* __restrict__ rst, const float* __restrict__ b,
                                 bf16* __restrict__ out, long long total, int HD) {
    long long i = (long long)blockIdx.x * blockDim.x + threadIdx.x;
    if (i >= total) return;
    float v = rst[i] + b[i % HD];
    out[i] = (bf16)(v > 0.f ? v : 0.f);
}

__global__ void k_bias_out(float* __restrict__ out, const float* __restrict__ b,
                           long long total, int C) {
    long long i = (long long)blockIdx.x * blockDim.x + threadIdx.x;
    if (i >= total) return;
    out[i] = out[i] + b[i % C];
}

// ----------------------------------------------------------------- launcher

static inline int nblk256(long long n) { return (int)((n + 255) / 256); }

extern "C" void kernel_launch(void* const* d_in, const int* in_sizes, int n_in,
                              void* d_out, int out_size, void* d_ws, size_t ws_size,
                              hipStream_t stream) {
    const int NN = 50000, E = 800000, F = 256;
    const int H1 = 4, D = 64, H2 = 1, C = 64;
    const int HD1 = H1 * D;     // 256
    const int HD2 = H2 * C;     // 64

    const float* x   = (const float*)d_in[0];
    const int* esrc  = (const int*)d_in[1];
    const int* edst  = (const int*)d_in[2];
    const float* W1  = (const float*)d_in[3];
    const float* al1 = (const float*)d_in[4];
    const float* ar1 = (const float*)d_in[5];
    const float* b1  = (const float*)d_in[6];
    const float* W2  = (const float*)d_in[7];
    const float* al2 = (const float*)d_in[8];
    const float* ar2 = (const float*)d_in[9];
    const float* b2  = (const float*)d_in[10];
    float* out = (float*)d_out;

    // workspace carve-up (256B aligned)
    char* base = (char*)d_ws;
    size_t off = 0;
    auto carve = [&](size_t bytes) -> char* {
        char* p = base + off;
        off = (off + bytes + 255) & ~(size_t)255;
        return p;
    };
    bf16*  xb     = (bf16*) carve((size_t)NN * F * 2);
    bf16*  Wp1    = (bf16*) carve((size_t)F * HD1 * 2);
    bf16*  Wp2    = (bf16*) carve((size_t)HD1 * HD2 * 2);
    float* feat1  = (float*)carve((size_t)NN * HD1 * 4);
    float* el1    = (float*)carve((size_t)NN * H1 * 4);
    float* er1    = (float*)carve((size_t)NN * H1 * 4);
    float* e1     = (float*)carve((size_t)E * H1 * 4);
    float* emax1  = (float*)carve((size_t)NN * H1 * 4);
    float* den1   = (float*)carve((size_t)NN * H1 * 4);
    float* rst1   = (float*)carve((size_t)NN * HD1 * 4);
    bf16*  h1b    = (bf16*) carve((size_t)NN * HD1 * 2);
    float* feat2  = (float*)carve((size_t)NN * HD2 * 4);
    float* el2    = (float*)carve((size_t)NN * H2 * 4);
    float* er2    = (float*)carve((size_t)NN * H2 * 4);
    float* e2     = (float*)carve((size_t)E * H2 * 4);
    float* emax2  = (float*)carve((size_t)NN * H2 * 4);
    float* den2   = (float*)carve((size_t)NN * H2 * 4);
    (void)ws_size; (void)in_sizes; (void)n_in; (void)out_size;

    // ---- init accumulators (every call: harness replays without re-poisoning)
    k_fill_f32<<<nblk256((long long)NN * HD1), 256, 0, stream>>>(rst1, 0.f, (long long)NN * HD1);
    k_fill_f32<<<nblk256((long long)NN * H1), 256, 0, stream>>>(emax1, -1e30f, (long long)NN * H1);
    k_fill_f32<<<nblk256((long long)NN * H1), 256, 0, stream>>>(den1, 0.f, (long long)NN * H1);
    k_fill_f32<<<nblk256((long long)NN * HD2), 256, 0, stream>>>(out, 0.f, (long long)NN * HD2);
    k_fill_f32<<<nblk256((long long)NN * H2), 256, 0, stream>>>(emax2, -1e30f, (long long)NN * H2);
    k_fill_f32<<<nblk256((long long)NN * H2), 256, 0, stream>>>(den2, 0.f, (long long)NN * H2);

    // ---- precision conversion + weight packing
    k_f32_to_bf16<<<nblk256((long long)NN * F), 256, 0, stream>>>(x, xb, (long long)NN * F);
    k_pack_w<<<nblk256((long long)F * HD1), 256, 0, stream>>>(W1, Wp1, F, HD1);
    k_pack_w<<<nblk256((long long)HD1 * HD2), 256, 0, stream>>>(W2, Wp2, HD1, HD2);

    // ---- layer 1: feat1 = x @ W1   (M=50000, N=256, K=256)
    {
        int tiles = (NN / 16) * (HD1 / 64);          // 12500 wave tiles
        int blocks = (tiles + 7) / 8;                // 8 waves / 256-thread block
        k_gemm_wmma_bf16<<<blocks, 256, 0, stream>>>(xb, Wp1, feat1, NN, HD1, F);
    }
    k_attn_dots<<<nblk256((long long)NN * H1), 256, 0, stream>>>(feat1, al1, ar1, el1, er1, NN, H1, D);
    k_edge_logits<<<nblk256((long long)E * H1), 256, 0, stream>>>(esrc, edst, el1, er1, e1, emax1, E, H1);
    k_edge_exp<<<nblk256((long long)E * H1), 256, 0, stream>>>(edst, emax1, e1, den1, E, H1);
    k_edge_scatter<<<nblk256((long long)E * HD1), 256, 0, stream>>>(esrc, edst, feat1, e1, den1, rst1, E, H1, D);
    k_bias_relu_bf16<<<nblk256((long long)NN * HD1), 256, 0, stream>>>(rst1, b1, h1b, (long long)NN * HD1, HD1);

    // ---- layer 2: feat2 = h1 @ W2  (M=50000, N=64, K=256)
    {
        int tiles = (NN / 16) * (HD2 / 64);          // 3125 wave tiles
        int blocks = (tiles + 7) / 8;
        k_gemm_wmma_bf16<<<blocks, 256, 0, stream>>>(h1b, Wp2, feat2, NN, HD2, HD1);
    }
    k_attn_dots<<<nblk256((long long)NN * H2), 256, 0, stream>>>(feat2, al2, ar2, el2, er2, NN, H2, C);
    k_edge_logits<<<nblk256((long long)E * H2), 256, 0, stream>>>(esrc, edst, el2, er2, e2, emax2, E, H2);
    k_edge_exp<<<nblk256((long long)E * H2), 256, 0, stream>>>(edst, emax2, e2, den2, E, H2);
    k_edge_scatter<<<nblk256((long long)E * HD2), 256, 0, stream>>>(esrc, edst, feat2, e2, den2, out, E, H2, C);
    k_bias_out<<<nblk256((long long)NN * HD2), 256, 0, stream>>>(out, b2, (long long)NN * HD2, C);
}
